// Model_31044023616105
// MI455X (gfx1250) — matrix operand order repaired and checked by reading, not compiled, not run on hardware
//
#include <hip/hip_runtime.h>
#include <hip/hip_bf16.h>
#include <hip/hip_fp16.h>

typedef __attribute__((ext_vector_type(16))) _Float16 v16h;
typedef __attribute__((ext_vector_type(8)))  float    v8f;

#define HID   128
#define BATCH 4
#define TLEN  16384
#define APAD  136   // Ash row stride in halves: 272B -> 4-bank rotation, no conflicts

__device__ __forceinline__ float sigmoidf_fast(float x) {
    return 1.0f / (1.0f + __expf(-x));
}
__device__ __forceinline__ float tanhf_fast(float x) {
    float e = __expf(2.0f * x);
    return (e - 1.0f) / (e + 1.0f);
}
__device__ __forceinline__ float eluf_fast(float x) {
    return x > 0.0f ? x : (__expf(x) - 1.0f);
}

// LDS-only workgroup barrier: waits DS ops only (NOT global storecnt), so the
// per-step global hs store never stalls the critical path.
__device__ __forceinline__ void wg_barrier_lds() {
    asm volatile("s_wait_dscnt 0x0\n\t"
                 "s_barrier_signal -1\n\t"
                 "s_barrier_wait -1" ::: "memory");
}

// Build a 16-half fragment from two 16-byte chunks (lo -> VGPR0-3, hi -> VGPR4-7).
__device__ __forceinline__ v16h ld_frag(const _Float16* lo, const _Float16* hi) {
    v16h r;
    *(float4*)&r         = *(const float4*)lo;
    *(((float4*)&r) + 1) = *(const float4*)hi;
    return r;
}

__device__ __forceinline__ v8f splat8(float x) {
    v8f r;
#pragma unroll
    for (int i = 0; i < 8; ++i) r[i] = x;
    return r;
}

// ---------------------------------------------------------------------------
// Weight repack: W_ih/W_hh -> Bw f16 [768][128] (row n = output col, k contig),
// W1 -> f16 [64][128], W2 -> f16 [64][64].
// ---------------------------------------------------------------------------
__global__ void prep_weights(const float* __restrict__ W_ih, const float* __restrict__ W_hh,
                             const float* __restrict__ W1,   const float* __restrict__ W2,
                             _Float16* __restrict__ Bw, _Float16* __restrict__ W1h,
                             _Float16* __restrict__ W2h) {
    int idx = blockIdx.x * blockDim.x + threadIdx.x;
    const int N_BW = 768 * 128, N_W1 = 64 * 128, N_W2 = 64 * 64;
    if (idx < N_BW) {
        int n = idx >> 7, k = idx & 127;
        float v = (n < 384) ? W_ih[n * 128 + k] : W_hh[(n - 384) * 128 + k];
        Bw[idx] = (_Float16)v;
    } else if (idx < N_BW + N_W1) {
        int i = idx - N_BW;
        W1h[i] = (_Float16)W1[i];
    } else if (idx < N_BW + N_W1 + N_W2) {
        int i = idx - N_BW - N_W1;
        W2h[i] = (_Float16)W2[i];
    }
}

// ---------------------------------------------------------------------------
// Embedding: xe[(t*4+b)*128 + j] = f16( W_embed[j,:] . x4 + b_embed[j] )
// ---------------------------------------------------------------------------
__global__ void embed_kernel(const float* __restrict__ px, const float* __restrict__ py,
                             const float* __restrict__ vx, const float* __restrict__ vy,
                             const float* __restrict__ W_embed, const float* __restrict__ b_embed,
                             _Float16* __restrict__ xe) {
    int j = threadIdx.x;  // 0..127
    float w0 = W_embed[j * 4 + 0], w1 = W_embed[j * 4 + 1];
    float w2 = W_embed[j * 4 + 2], w3 = W_embed[j * 4 + 3];
    float bb = b_embed[j];
    for (int r = blockIdx.x; r < TLEN * BATCH; r += gridDim.x) {
        int t = r >> 2, b = r & 3;
        int s = b * TLEN + t;
        float acc = bb + w0 * px[s] + w1 * py[s] + w2 * vx[s] + w3 * vy[s];
        xe[r * HID + j] = (_Float16)acc;
    }
}

// ---------------------------------------------------------------------------
// Serial GRU scan: one workgroup, 512 threads = 16 waves.
// Per step: one [16,128]x[128,768] WMMA GEMM.
//   A rows 0-3 = xe[t] (always), rows 4-7 = h.
//   For n<384:  C rows 0-3 = xe.W_ih, rows 4-7 = h.W_ih  -> gi picked by use_x.
//   For n>=384: C rows 4-7 = h.W_hh                      -> gh.
// B fragments stay resident in VGPRs for all 16384 steps; xe[t+1] is register-
// pipelined; f32 hidden state lives in a per-thread register.
// ---------------------------------------------------------------------------
__global__ void __launch_bounds__(512)
gru_scan(const _Float16* __restrict__ xe, const _Float16* __restrict__ Bw,
         const float* __restrict__ b_ih, const float* __restrict__ b_hh,
         const int* __restrict__ step_mask, const int* __restrict__ ctxp,
         _Float16* __restrict__ hs) {
    __shared__ __align__(16) _Float16 Ash[16][APAD];  // padded A tile (~4.3 KB)
    __shared__ float giS[BATCH][3 * HID];             // 6 KB
    __shared__ float ghS[BATCH][3 * HID];             // 6 KB
    __shared__ unsigned uxw[TLEN / 32];               // 2 KB use_x bitmask

    const int tid  = threadIdx.x;
    const int lane = tid & 31;
    const int w    = tid >> 5;    // wave 0..15
    const int ln   = lane & 15;
    const int g    = lane >> 4;

    int ctx = ctxp[0]; if (ctx < 1) ctx = 1;

    // Precompute use_x bitmask into LDS.
    for (int word = tid; word < TLEN / 32; word += 512) {
        unsigned bits = 0;
        int base = word * 32;
#pragma unroll 4
        for (int i = 0; i < 32; ++i) {
            int t = base + i;
            bits |= ((t < ctx) || (step_mask[t] == 0)) ? (1u << i) : 0u;
        }
        uxw[word] = bits;
    }

    // Init A (rows 0-3 = xe[0], rows 4-15 = 0).
    {
        int b = tid >> 7, j = tid & 127;
        Ash[4 + b][j]  = (_Float16)0.0f;
        Ash[8 + b][j]  = (_Float16)0.0f;
        Ash[12 + b][j] = (_Float16)0.0f;
        Ash[b][j]      = xe[b * HID + j];   // t = 0 always uses x
    }

    // Resident B fragments (3 N-tiles x 4 K-chunks) + bias splat values.
    v16h bF[3][4];
    float biasv[3];
    const int nbase = w * 48;
#pragma unroll
    for (int tt = 0; tt < 3; ++tt) {
        int n0 = nbase + tt * 16;
        const _Float16* brow = Bw + (n0 + ln) * HID;
#pragma unroll
        for (int kc = 0; kc < 4; ++kc) {
            const _Float16* p = brow + kc * 32 + g * 8;  // two 16B K slices: 8g..8g+7, 16+8g..16+8g+7
            bF[tt][kc] = ld_frag(p, p + 16);
        }
        biasv[tt] = (n0 < 384) ? b_ih[n0 + ln] : b_hh[n0 - 384 + ln];
    }

    __syncthreads();

    const int bb = tid >> 7;   // batch row for gate math
    const int jj = tid & 127;  // hidden index

    float hreg = 0.0f;                                // h[bb][jj], thread-private
    _Float16 xen = xe[(size_t)(BATCH * HID) + tid];   // xe for t = 1

    for (int t = 0; t < TLEN; ++t) {
        // Prefetch xe three steps ahead to keep the L2->register pipe warm.
        if (t + 3 < TLEN)
            __builtin_prefetch(xe + (size_t)(t + 3) * (BATCH * HID) + tid, 0, 1);

        // A fragments: 8x ds_load_b128 (conflict-free thanks to APAD), then tie
        // them as asm operands so the WMMAs cannot be hoisted above the loads
        // and the loads cannot sink below -> single dscnt wait per step.
        v16h aF0, aF1, aF2, aF3;
        {
            const _Float16* ar = &Ash[ln][g * 8];
            aF0 = ld_frag(ar,       ar + 16);
            aF1 = ld_frag(ar + 32,  ar + 48);
            aF2 = ld_frag(ar + 64,  ar + 80);
            aF3 = ld_frag(ar + 96,  ar + 112);
        }
        asm volatile("" : "+v"(aF0), "+v"(aF1), "+v"(aF2), "+v"(aF3));

        v8f c0 = splat8(biasv[0]);
        v8f c1 = splat8(biasv[1]);
        v8f c2 = splat8(biasv[2]);
        c0 = __builtin_amdgcn_wmma_f32_16x16x32_f16(false, aF0, false, bF[0][0], (short)0, c0, false, false);
        c1 = __builtin_amdgcn_wmma_f32_16x16x32_f16(false, aF0, false, bF[1][0], (short)0, c1, false, false);
        c2 = __builtin_amdgcn_wmma_f32_16x16x32_f16(false, aF0, false, bF[2][0], (short)0, c2, false, false);
        c0 = __builtin_amdgcn_wmma_f32_16x16x32_f16(false, aF1, false, bF[0][1], (short)0, c0, false, false);
        c1 = __builtin_amdgcn_wmma_f32_16x16x32_f16(false, aF1, false, bF[1][1], (short)0, c1, false, false);
        c2 = __builtin_amdgcn_wmma_f32_16x16x32_f16(false, aF1, false, bF[2][1], (short)0, c2, false, false);
        c0 = __builtin_amdgcn_wmma_f32_16x16x32_f16(false, aF2, false, bF[0][2], (short)0, c0, false, false);
        c1 = __builtin_amdgcn_wmma_f32_16x16x32_f16(false, aF2, false, bF[1][2], (short)0, c1, false, false);
        c2 = __builtin_amdgcn_wmma_f32_16x16x32_f16(false, aF2, false, bF[2][2], (short)0, c2, false, false);
        c0 = __builtin_amdgcn_wmma_f32_16x16x32_f16(false, aF3, false, bF[0][3], (short)0, c0, false, false);
        c1 = __builtin_amdgcn_wmma_f32_16x16x32_f16(false, aF3, false, bF[1][3], (short)0, c1, false, false);
        c2 = __builtin_amdgcn_wmma_f32_16x16x32_f16(false, aF3, false, bF[2][3], (short)0, c2, false, false);

        // Scatter C to LDS.  use_x picks which C rows feed gi (uniform select).
        bool sux = (uxw[t >> 5] >> (t & 31)) & 1u;
        if (lane < 16) {
            if (w < 8) {
#pragma unroll
                for (int r = 0; r < 4; ++r) {
                    giS[r][nbase +  0 + ln] = sux ? c0[r] : c0[r + 4];
                    giS[r][nbase + 16 + ln] = sux ? c1[r] : c1[r + 4];
                    giS[r][nbase + 32 + ln] = sux ? c2[r] : c2[r + 4];
                }
            } else {
                int nb = nbase - 384;
#pragma unroll
                for (int r = 0; r < 4; ++r) {
                    ghS[r][nb +  0 + ln] = c0[r + 4];
                    ghS[r][nb + 16 + ln] = c1[r + 4];
                    ghS[r][nb + 32 + ln] = c2[r + 4];
                }
            }
        }
        wg_barrier_lds();

        // All waves have consumed A -> safe to drop in next step's input rows.
        if (t + 1 < TLEN)
            Ash[bb][jj] = xen;                    // rows 0-3 = xe[t+1]

        // Issue the t+2 input load now; latency hides behind gate math.
        _Float16 xen_next = (_Float16)0.0f;
        if (t + 2 < TLEN)
            xen_next = xe[(size_t)(t + 2) * (BATCH * HID) + tid];

        // Gate math: one thread per (b, j).
        float i_r = giS[bb][jj];
        float i_z = giS[bb][HID + jj];
        float i_n = giS[bb][2 * HID + jj];
        float h_r = ghS[bb][jj];
        float h_z = ghS[bb][HID + jj];
        float h_n = ghS[bb][2 * HID + jj];
        float rg = sigmoidf_fast(i_r + h_r);
        float zg = sigmoidf_fast(i_z + h_z);
        float ng = tanhf_fast(i_n + rg * h_n);
        float hn = (1.0f - zg) * ng + zg * hreg;

        hreg = hn;
        _Float16 hh = (_Float16)hn;
        Ash[4 + bb][jj] = hh;                     // rows 4-7 = h for next step
        hs[((size_t)t * BATCH + bb) * HID + jj] = hh;

        xen = xen_next;
        wg_barrier_lds();
    }
}

// ---------------------------------------------------------------------------
// MLP head: per block 128 rows of hs. L1/L2 via WMMA (ELU fused), L3 via VALU.
// LDS rows padded (72 / 68 elems) to avoid bank conflicts.
// ---------------------------------------------------------------------------
__global__ void __launch_bounds__(256)
head_kernel(const _Float16* __restrict__ hs,
            const _Float16* __restrict__ W1h, const float* __restrict__ b1,
            const _Float16* __restrict__ W2h, const float* __restrict__ b2,
            const float* __restrict__ W3, const float* __restrict__ b3,
            float* __restrict__ out) {
    __shared__ __align__(16) _Float16 y1S[128][72];   // 18 KB, 144B rows
    __shared__ float y2S[128][68];                    // 34 KB, 272B rows

    const int tid  = threadIdx.x;
    const int lane = tid & 31;
    const int w    = tid >> 5;    // wave 0..7 (one M-tile each)
    const int ln   = lane & 15;
    const int g    = lane >> 4;
    const int rowBase = blockIdx.x * 128;

    // ---- layer 1: [16,128] x [128,64] per wave ----
    const int m = rowBase + w * 16 + ln;
    v8f c[4];
#pragma unroll
    for (int nt = 0; nt < 4; ++nt) c[nt] = splat8(b1[nt * 16 + ln]);

#pragma unroll
    for (int kc = 0; kc < 4; ++kc) {
        const _Float16* ar = hs + (size_t)m * 128 + kc * 32 + g * 8;
        v16h aF = ld_frag(ar, ar + 16);
#pragma unroll
        for (int nt = 0; nt < 4; ++nt) {
            const _Float16* br = W1h + (nt * 16 + ln) * 128 + kc * 32 + g * 8;
            v16h bFr = ld_frag(br, br + 16);
            c[nt] = __builtin_amdgcn_wmma_f32_16x16x32_f16(false, aF, false, bFr, (short)0, c[nt], false, false);
        }
    }
    // ELU + f16 into LDS for layer 2's A operand.
#pragma unroll
    for (int nt = 0; nt < 4; ++nt) {
#pragma unroll
        for (int r = 0; r < 8; ++r) {
            int ml = (g == 0) ? r : (8 + r);
            y1S[w * 16 + ml][nt * 16 + ln] = (_Float16)eluf_fast(c[nt][r]);
        }
    }
    __syncthreads();

    // ---- layer 2: [16,64] x [64,64] per wave ----
    v8f d[4];
#pragma unroll
    for (int nt = 0; nt < 4; ++nt) d[nt] = splat8(b2[nt * 16 + ln]);
#pragma unroll
    for (int kc = 0; kc < 2; ++kc) {
        const _Float16* ar = &y1S[w * 16 + ln][kc * 32 + g * 8];
        v16h aF = ld_frag(ar, ar + 16);
#pragma unroll
        for (int nt = 0; nt < 4; ++nt) {
            const _Float16* br = W2h + (nt * 16 + ln) * 64 + kc * 32 + g * 8;
            v16h bFr = ld_frag(br, br + 16);
            d[nt] = __builtin_amdgcn_wmma_f32_16x16x32_f16(false, aF, false, bFr, (short)0, d[nt], false, false);
        }
    }
#pragma unroll
    for (int nt = 0; nt < 4; ++nt) {
#pragma unroll
        for (int r = 0; r < 8; ++r) {
            int ml = (g == 0) ? r : (8 + r);
            y2S[w * 16 + ml][nt * 16 + ln] = eluf_fast(d[nt][r]);
        }
    }
    __syncthreads();

    // ---- layer 3: [128,64] x [64,2] via VALU; thread = (row, out-col) ----
    const int rloc = tid >> 1;
    const int o    = tid & 1;
    float acc = b3[o];
    const float* w3 = W3 + o * 64;
#pragma unroll 8
    for (int k = 0; k < 64; ++k) acc += y2S[rloc][k] * w3[k];
    int mg = rowBase + rloc;
    int t = mg >> 2, b = mg & 3;
    out[(size_t)o * (BATCH * TLEN) + (size_t)b * TLEN + t] = acc;
}

// ---------------------------------------------------------------------------
extern "C" void kernel_launch(void* const* d_in, const int* in_sizes, int n_in,
                              void* d_out, int out_size, void* d_ws, size_t ws_size,
                              hipStream_t stream) {
    (void)in_sizes; (void)n_in; (void)out_size; (void)ws_size;
    const float* px      = (const float*)d_in[0];
    const float* py      = (const float*)d_in[1];
    const float* vx      = (const float*)d_in[2];
    const float* vy      = (const float*)d_in[3];
    const float* W_embed = (const float*)d_in[4];
    const float* b_embed = (const float*)d_in[5];
    const float* W_ih    = (const float*)d_in[6];
    const float* W_hh    = (const float*)d_in[7];
    const float* b_ih    = (const float*)d_in[8];
    const float* b_hh    = (const float*)d_in[9];
    const float* W1      = (const float*)d_in[10];
    const float* b1      = (const float*)d_in[11];
    const float* W2      = (const float*)d_in[12];
    const float* b2      = (const float*)d_in[13];
    const float* W3      = (const float*)d_in[14];
    const float* b3      = (const float*)d_in[15];
    const int* step_mask = (const int*)d_in[16];
    const int* ctx       = (const int*)d_in[17];

    char* ws = (char*)d_ws;
    _Float16* xe  = (_Float16*)(ws);                                   // 16 MB
    _Float16* hsb = (_Float16*)(ws + (size_t)16 * 1024 * 1024);        // 16 MB
    _Float16* Bw  = (_Float16*)(ws + (size_t)32 * 1024 * 1024);        // 192 KB
    _Float16* W1h = (_Float16*)(ws + (size_t)32 * 1024 * 1024 + 768 * 128 * 2);
    _Float16* W2h = (_Float16*)(ws + (size_t)32 * 1024 * 1024 + 768 * 128 * 2 + 64 * 128 * 2);

    const int prep_total = 768 * 128 + 64 * 128 + 64 * 64;
    prep_weights<<<(prep_total + 255) / 256, 256, 0, stream>>>(W_ih, W_hh, W1, W2, Bw, W1h, W2h);
    embed_kernel<<<2048, 128, 0, stream>>>(px, py, vx, vy, W_embed, b_embed, xe);
    gru_scan<<<1, 512, 0, stream>>>(xe, Bw, b_ih, b_hh, step_mask, ctx, hsb);
    head_kernel<<<(TLEN * BATCH) / 128, 256, 0, stream>>>(hsb, W1h, b1, W2h, b2, W3, b3, (float*)d_out);
}
